// GroupedAttention_24678882083031
// MI455X (gfx1250) — matrix operand order repaired and checked by reading, not compiled, not run on hardware
//
#include <hip/hip_runtime.h>

// ---------- types ----------
typedef __attribute__((ext_vector_type(16))) __bf16 v16bf;
typedef __attribute__((ext_vector_type(8)))  float  v8f;
typedef __attribute__((ext_vector_type(4)))  unsigned int u32x4;

union FragAB {
  v16bf v;
  u32x4 q[2];
  unsigned short u[16];
};

__device__ __forceinline__ unsigned short f2bf(float f) {
  union { float f; unsigned int u; } c; c.f = f;
  unsigned int u = c.u;
  unsigned int r = (u + 0x7FFFu + ((u >> 16) & 1u)) >> 16;
  return (unsigned short)r;
}

__device__ __forceinline__ v8f vzero8() {
  v8f z;
#pragma unroll
  for (int i = 0; i < 8; ++i) z[i] = 0.0f;
  return z;
}

// async global->LDS copy of 32 bytes (two b128 ops sharing one LDS base VGPR;
// the instruction offset applies to BOTH the global and LDS address)
__device__ __forceinline__ void async_copy32(const unsigned short* gsrc,
                                             unsigned short* lds_dst) {
  unsigned loff = (unsigned)(size_t)lds_dst;  // low 32 bits of generic = LDS offset
  unsigned long long ga = (unsigned long long)(size_t)gsrc;
  asm volatile("global_load_async_to_lds_b128 %0, %1, off"
               :: "v"(loff), "v"(ga) : "memory");
  asm volatile("global_load_async_to_lds_b128 %0, %1, off offset:16"
               :: "v"(loff), "v"(ga) : "memory");
}

// ---------- f32 -> bf16 cast (row-major preserved) ----------
__global__ void cast_f32_bf16(const float* __restrict__ in,
                              unsigned short* __restrict__ out, int n) {
  int i = blockIdx.x * blockDim.x + threadIdx.x;
  int stride = gridDim.x * blockDim.x;
  for (; i < n; i += stride) out[i] = f2bf(in[i]);
}

// ---------- f32 [R,C] -> bf16 [C,R] tiled transpose (one-time, for weights) ----------
__global__ void cast_transpose(const float* __restrict__ in,
                               unsigned short* __restrict__ out,
                               int R, int C) {
  __shared__ unsigned short tile[32][33];
  int tc = blockIdx.x * 32;
  int tr = blockIdx.y * 32;
  int lx = threadIdx.x & 31;
  int ly = threadIdx.x >> 5;
#pragma unroll
  for (int p = 0; p < 4; ++p) {
    int r = tr + ly + p * 8, c = tc + lx;
    tile[ly + p * 8][lx] = f2bf(in[(size_t)r * C + c]);
  }
  __syncthreads();
#pragma unroll
  for (int p = 0; p < 4; ++p) {
    int c = tc + ly + p * 8, r = tr + lx;
    out[(size_t)c * R + r] = tile[lx][ly + p * 8];
  }
}

// ---------- V transpose: f32 [B,S,NKV,128] -> bf16 [B,NKV,128,S] ----------
__global__ void cast_transpose_v(const float* __restrict__ vf,
                                 unsigned short* __restrict__ vt, int S_) {
  const int NKV = 4, D = 128;
  __shared__ unsigned short tile[32][33];
  int bh = blockIdx.z;
  int s0 = blockIdx.x * 32;
  int d0 = blockIdx.y * 32;
  int lx = threadIdx.x & 31;
  int ly = threadIdx.x >> 5;
  const float* src = vf + ((size_t)(bh / NKV) * S_ * NKV + (bh % NKV)) * D;
#pragma unroll
  for (int p = 0; p < 4; ++p) {
    int s = s0 + ly + p * 8, d = d0 + lx;
    tile[ly + p * 8][lx] = f2bf(src[(size_t)s * NKV * D + d]);
  }
  __syncthreads();
  unsigned short* dst = vt + (size_t)bh * D * S_;
#pragma unroll
  for (int p = 0; p < 4; ++p) {
    int d = d0 + ly + p * 8, s = s0 + lx;
    dst[(size_t)d * S_ + s] = tile[lx][ly + p * 8];
  }
}

// ---------- bf16 TN GEMM: C[M,N] = A[M,K] * BT[N,K]^T, f32 out ----------
// block tile 128(M) x 128(N), 8 waves (4x2), wave tile 32x64, K-step 32.
// Double-buffered LDS, async global->LDS DMA (ASYNCcnt), 8 WMMA/wave/K-step.
// Last iteration peeled so the steady-state loop is branch-free.
__launch_bounds__(256)
__global__ void gemm_bf16_tn(const unsigned short* __restrict__ A,
                             const unsigned short* __restrict__ BT,
                             float* __restrict__ C,
                             int M, int N, int K) {
  __shared__ unsigned short Alds[2][128 * 32];  // [m][k]
  __shared__ unsigned short Blds[2][128 * 32];  // [n][k]

  const int tid  = threadIdx.x;
  const int lane = tid & 31;
  const int wave = tid >> 5;
  const int wm = wave >> 1;   // 0..3  (rows: wm*32)
  const int wn = wave & 1;    // 0..1  (cols: wn*64)
  const int mBase = blockIdx.y * 128;
  const int nBase = blockIdx.x * 128;
  const int l15  = lane & 15;
  const int hsel = (lane < 16) ? 0 : 1;

  // staging: each thread owns (row = tid/2, col = (tid&1)*16), 16 elems of A and B
  const int srow = tid >> 1;
  const int scol = (tid & 1) * 16;
  const unsigned short* srcA = A  + (size_t)(mBase + srow) * K + scol;
  const unsigned short* srcB = BT + (size_t)(nBase + srow) * K + scol;
  unsigned short* dstA0 = &Alds[0][srow * 32 + scol];
  unsigned short* dstA1 = &Alds[1][srow * 32 + scol];
  unsigned short* dstB0 = &Blds[0][srow * 32 + scol];
  unsigned short* dstB1 = &Blds[1][srow * 32 + scol];

  v8f acc[2][4];
#pragma unroll
  for (int i = 0; i < 2; ++i)
#pragma unroll
    for (int j = 0; j < 4; ++j) acc[i][j] = vzero8();

  // compute on buffer `cur` (shared by steady-state loop and epilogue)
  auto computeTile = [&](int cur) {
    const unsigned short* Ab = Alds[cur];
    const unsigned short* Bb = Blds[cur];
    FragAB a[2], b[4];
#pragma unroll
    for (int i = 0; i < 2; ++i) {
      int m  = wm * 32 + i * 16 + l15;
      int kb = hsel * 8;
      a[i].q[0] = *(const u32x4*)&Ab[m * 32 + kb];
      a[i].q[1] = *(const u32x4*)&Ab[m * 32 + 16 + kb];
    }
#pragma unroll
    for (int j = 0; j < 4; ++j) {
      int n  = wn * 64 + j * 16 + l15;
      int kb = hsel * 8;
      b[j].q[0] = *(const u32x4*)&Bb[n * 32 + kb];
      b[j].q[1] = *(const u32x4*)&Bb[n * 32 + kb + 16];
    }
#pragma unroll
    for (int i = 0; i < 2; ++i)
#pragma unroll
      for (int j = 0; j < 4; ++j)
        acc[i][j] = __builtin_amdgcn_wmma_f32_16x16x32_bf16(
            false, a[i].v, false, b[j].v, (short)0, acc[i][j], false, false);
  };

  // prologue: async-stage tile 0 into buffer 0
  async_copy32(srcA, dstA0);
  async_copy32(srcB, dstB0);

  const int nsteps = K >> 5;  // >= 2 for all our shapes
  for (int s = 0; s < nsteps - 1; ++s) {
    const int cur = s & 1;
    const int k0 = s << 5;
    // issue next tile into the other buffer (branch-free steady state)
    async_copy32(srcA + k0 + 32, cur ? dstA0 : dstA1);
    async_copy32(srcB + k0 + 32, cur ? dstB0 : dstB1);
    // speculative L2 prefetch two tiles ahead (OOB prefetch is dropped by HW)
    __builtin_prefetch(srcA + k0 + 64, 0, 1);
    __builtin_prefetch(srcB + k0 + 64, 0, 1);
    asm volatile("s_wait_asynccnt 0x4" ::: "memory");  // current tile landed
    __syncthreads();
    computeTile(cur);
    __syncthreads();  // all frag reads done before buffer reuse
  }
  // epilogue: last tile
  asm volatile("s_wait_asynccnt 0x0" ::: "memory");
  __syncthreads();
  computeTile((nsteps - 1) & 1);

#pragma unroll
  for (int i = 0; i < 2; ++i)
#pragma unroll
    for (int j = 0; j < 4; ++j)
#pragma unroll
      for (int r = 0; r < 8; ++r) {
        int m = mBase + wm * 32 + i * 16 + r + hsel * 8;
        int n = nBase + wn * 64 + j * 16 + l15;
        C[(size_t)m * N + n] = acc[i][j][r];
      }
}

// ---------- per-head RMSNorm + RoPE, f32 in -> bf16 out ----------
__global__ void norm_rope(const float* __restrict__ in,
                          const float* __restrict__ scale,
                          const float* __restrict__ cosp,   // [B*S,128]
                          const float* __restrict__ sinp,
                          unsigned short* __restrict__ out,
                          int nheads) {
  int head = blockIdx.x % nheads;
  int bs   = blockIdx.x / nheads;
  int d    = threadIdx.x;
  size_t base = ((size_t)bs * nheads + head) * 128;
  float x = in[base + d];

  __shared__ float red[128];
  __shared__ float sh[128];
  red[d] = x * x;
  __syncthreads();
  for (int off = 64; off > 0; off >>= 1) {
    if (d < off) red[d] += red[d + off];
    __syncthreads();
  }
  float rn = rsqrtf(red[0] * (1.0f / 128.0f) + 1e-6f);
  float xn = x * rn * scale[d];
  sh[d] = xn;
  __syncthreads();
  float partner = sh[d ^ 64];
  float rot = (d < 64) ? -partner : partner;
  float c = cosp[(size_t)bs * 128 + d];
  float s = sinp[(size_t)bs * 128 + d];
  out[base + d] = f2bf(xn * c + rot * s);
}

// ---------- flash attention ----------
// block: one (b,h), 128 q-rows, 8 waves x 16 q-rows. Key step 32.
// V pre-transposed [B,NKV,128,S]; K/V tiles staged via async global->LDS DMA.
__launch_bounds__(256)
__global__ void attn_fwd(const unsigned short* __restrict__ Q,  // [B,S,16,128]
                         const unsigned short* __restrict__ Kt, // [B,S,4,128]
                         const unsigned short* __restrict__ Vt, // [B,4,128,S]
                         const float* __restrict__ mask,        // [B,1,S,S]
                         unsigned short* __restrict__ O,        // [B,S,16,128]
                         int Bc, int Sc) {
  const int NH = 16, NKV = 4, D = 128;
  const int blocksPerBH = Sc / 128;
  int bh   = blockIdx.x / blocksPerBH;
  int qblk = blockIdx.x % blocksPerBH;
  int b = bh / NH;
  int h = bh % NH;
  int hk = h / (NH / NKV);
  int q0 = qblk * 128;

  const int tid  = threadIdx.x;
  const int lane = tid & 31;
  const int wave = tid >> 5;
  const int l15  = lane & 15;
  const int hsel = (lane < 16) ? 0 : 1;
  const int qr   = q0 + wave * 16;

  __shared__ unsigned short Klds[32 * 128];   // [key][d]
  __shared__ unsigned short Vlds[128 * 32];   // [d][key]
  __shared__ unsigned short Plds[8][16 * 32]; // per-wave P tile [row][key]

  // resident Q fragments: 4 chunks of K(d)=32
  FragAB qa[4];
  {
    size_t rowBase = (((size_t)(b * Sc + qr + l15) * NH) + h) * D;
    int kb = hsel * 8;
#pragma unroll
    for (int c = 0; c < 4; ++c) {
      qa[c].q[0] = *(const u32x4*)&Q[rowBase + c * 32 + kb];
      qa[c].q[1] = *(const u32x4*)&Q[rowBase + c * 32 + 16 + kb];
    }
  }

  // staging decompositions (both cover 32x128 = 4096 elems, 16/thread)
  const int kKr = (tid * 16) >> 7;          // K: key row 0..31
  const int kDc = (tid * 16) & 127;         // K: d col (multiple of 16)
  const int vDr = (tid * 16) >> 5;          // V: d row 0..127
  const int vKc = (tid * 16) & 31;          // V: key col 0 or 16
  const unsigned short* kSrc =
      Kt + (((size_t)(b * Sc) + kKr) * NKV + hk) * D + kDc;
  const unsigned short* vSrc =
      Vt + ((size_t)(b * NKV + hk) * D + vDr) * Sc + vKc;
  unsigned short* kDst = &Klds[kKr * D + kDc];
  unsigned short* vDst = &Vlds[vDr * 32 + vKc];

  v8f o[8];
#pragma unroll
  for (int c = 0; c < 8; ++c) o[c] = vzero8();
  float mrow[8], lrow[8];
#pragma unroll
  for (int r = 0; r < 8; ++r) { mrow[r] = -INFINITY; lrow[r] = 0.0f; }

  const float scl = 0.08838834764831845f;  // 128^-0.5

  for (int k0 = 0; k0 < Sc; k0 += 32) {
    // ---- stage K and V tiles via async DMA ----
    {
      const unsigned short* ks = kSrc + (size_t)k0 * NKV * D;
      const unsigned short* vs = vSrc + k0;
      async_copy32(ks, kDst);
      async_copy32(vs, vDst);
      // speculative prefetch of next tile (dropped by HW if OOB)
      __builtin_prefetch(ks + (size_t)32 * NKV * D, 0, 1);
      __builtin_prefetch(vs + 32, 0, 1);
    }
    asm volatile("s_wait_asynccnt 0x0" ::: "memory");
    __syncthreads();

    // ---- logits: S[16 x 32] = Q(16x128) . K^T ----
    v8f sfr[2];
#pragma unroll
    for (int f = 0; f < 2; ++f) {
      v8f acc = vzero8();
      int key = f * 16 + l15;
      int kb  = hsel * 8;
#pragma unroll
      for (int c = 0; c < 4; ++c) {
        FragAB bk;
        bk.q[0] = *(const u32x4*)&Klds[key * D + c * 32 + kb];
        bk.q[1] = *(const u32x4*)&Klds[key * D + c * 32 + kb + 16];
        acc = __builtin_amdgcn_wmma_f32_16x16x32_bf16(
            false, qa[c].v, false, bk.v, (short)0, acc, false, false);
      }
      sfr[f] = acc;
    }

    // ---- online softmax (accumulator reg r <-> q row qr + r + hsel*8) ----
#pragma unroll
    for (int r = 0; r < 8; ++r) {
      int qrow = qr + r + hsel * 8;
      const float* mrow_ptr = mask + ((size_t)b * Sc + qrow) * Sc + k0;
      float s0 = sfr[0][r] * scl + mrow_ptr[l15];
      float s1 = sfr[1][r] * scl + mrow_ptr[16 + l15];
      float mx = fmaxf(s0, s1);
#pragma unroll
      for (int msk = 1; msk < 16; msk <<= 1)
        mx = fmaxf(mx, __shfl_xor(mx, msk, 32));
      float mnew  = fmaxf(mrow[r], mx);
      float alpha = __expf(mrow[r] - mnew);
      float p0 = __expf(s0 - mnew);
      float p1 = __expf(s1 - mnew);
      float ps = p0 + p1;
#pragma unroll
      for (int msk = 1; msk < 16; msk <<= 1)
        ps += __shfl_xor(ps, msk, 32);
      lrow[r] = lrow[r] * alpha + ps;
      mrow[r] = mnew;
#pragma unroll
      for (int c = 0; c < 8; ++c) o[c][r] *= alpha;
      int prow = r + hsel * 8;
      Plds[wave][prow * 32 + l15]      = f2bf(p0);
      Plds[wave][prow * 32 + 16 + l15] = f2bf(p1);
    }
    __syncthreads();  // P visible + dscnt drained

    // ---- O += P(16x32) . V(32x128) ----
    FragAB pa;
    {
      int kb = hsel * 8;
      pa.q[0] = *(const u32x4*)&Plds[wave][l15 * 32 + kb];
      pa.q[1] = *(const u32x4*)&Plds[wave][l15 * 32 + 16 + kb];
    }
#pragma unroll
    for (int c = 0; c < 8; ++c) {
      FragAB vb;
      int n  = c * 16 + l15;
      int kb = hsel * 8;
      vb.q[0] = *(const u32x4*)&Vlds[n * 32 + kb];
      vb.q[1] = *(const u32x4*)&Vlds[n * 32 + kb + 16];
      o[c] = __builtin_amdgcn_wmma_f32_16x16x32_bf16(
          false, pa.v, false, vb.v, (short)0, o[c], false, false);
    }
    __syncthreads();  // protect K/V LDS for next tile
  }

  // ---- epilogue: divide by l, store bf16 [B,S,NH,D] ----
#pragma unroll
  for (int c = 0; c < 8; ++c)
#pragma unroll
    for (int r = 0; r < 8; ++r) {
      int qrow = qr + r + hsel * 8;
      size_t idx = (((size_t)(b * Sc + qrow) * NH) + h) * D + c * 16 + l15;
      O[idx] = f2bf(o[c][r] / lrow[r]);
    }
}

// ---------- host launch ----------
extern "C" void kernel_launch(void* const* d_in, const int* in_sizes, int n_in,
                              void* d_out, int out_size, void* d_ws, size_t ws_size,
                              hipStream_t stream) {
  (void)in_sizes; (void)n_in; (void)out_size; (void)ws_size;
  const float* x     = (const float*)d_in[0];
  const float* mask  = (const float*)d_in[1];
  const float* cosp  = (const float*)d_in[2];
  const float* sinp  = (const float*)d_in[3];
  const float* Wq    = (const float*)d_in[4];
  const float* Wk    = (const float*)d_in[5];
  const float* Wv    = (const float*)d_in[6];
  const float* Wo    = (const float*)d_in[7];
  const float* qscal = (const float*)d_in[8];
  const float* kscal = (const float*)d_in[9];
  float* out = (float*)d_out;

  const int B = 2, S = 2048, H = 2048, NH = 16, NKV = 4, D = 128;
  const size_t BS  = (size_t)B * S;    // 4096
  const size_t KVW = (size_t)NKV * D;  // 512

  char* ws = (char*)d_ws;
  size_t off = 0;
  auto alloc = [&](size_t bytes) -> void* {
    void* p = ws + off;
    off += (bytes + 255) & ~(size_t)255;
    return p;
  };
  unsigned short* xb   = (unsigned short*)alloc(BS * H * 2);           // x bf16
  unsigned short* wqT  = (unsigned short*)alloc((size_t)H * H * 2);    // Wq^T
  unsigned short* wkT  = (unsigned short*)alloc((size_t)H * KVW * 2);  // Wk^T
  unsigned short* wvT  = (unsigned short*)alloc((size_t)H * KVW * 2);  // Wv^T
  unsigned short* woT  = (unsigned short*)alloc((size_t)H * H * 2);    // Wo^T
  float* qf = (float*)alloc(BS * H * 4);
  float* kf = (float*)alloc(BS * KVW * 4);
  float* vf = (float*)alloc(BS * KVW * 4);
  unsigned short* qb   = (unsigned short*)alloc(BS * H * 2);
  unsigned short* kbuf = (unsigned short*)alloc(BS * KVW * 2);
  unsigned short* vt   = (unsigned short*)alloc(BS * KVW * 2);  // [B,NKV,D,S]
  unsigned short* ao   = (unsigned short*)alloc(BS * H * 2);

  // 1) casts: x stays row-major; weights are cast+transposed once
  cast_f32_bf16<<<4096, 256, 0, stream>>>(x, xb, (int)(BS * H));
  cast_transpose<<<dim3(H / 32, H / 32), 256, 0, stream>>>(Wq, wqT, H, H);
  cast_transpose<<<dim3((int)(KVW / 32), H / 32), 256, 0, stream>>>(Wk, wkT, H, (int)KVW);
  cast_transpose<<<dim3((int)(KVW / 32), H / 32), 256, 0, stream>>>(Wv, wvT, H, (int)KVW);
  cast_transpose<<<dim3(H / 32, H / 32), 256, 0, stream>>>(Wo, woT, H, H);

  // 2) QKV projections (TN bf16 WMMA GEMM, f32 out)
  gemm_bf16_tn<<<dim3(H / 128, (int)(BS / 128)), 256, 0, stream>>>(
      xb, wqT, qf, (int)BS, H, H);
  gemm_bf16_tn<<<dim3((int)(KVW / 128), (int)(BS / 128)), 256, 0, stream>>>(
      xb, wkT, kf, (int)BS, (int)KVW, H);
  gemm_bf16_tn<<<dim3((int)(KVW / 128), (int)(BS / 128)), 256, 0, stream>>>(
      xb, wvT, vf, (int)BS, (int)KVW, H);

  // 3) per-head RMSNorm + RoPE for q,k; transpose-cast for v
  norm_rope<<<(int)(BS * NH), 128, 0, stream>>>(qf, qscal, cosp, sinp, qb, NH);
  norm_rope<<<(int)(BS * NKV), 128, 0, stream>>>(kf, kscal, cosp, sinp, kbuf, NKV);
  cast_transpose_v<<<dim3(S / 32, D / 32, B * NKV), 256, 0, stream>>>(vf, vt, S);

  // 4) flash attention
  attn_fwd<<<B * NH * (S / 128), 256, 0, stream>>>(qb, kbuf, vt, mask, ao, B, S);

  // 5) output projection -> f32
  gemm_bf16_tn<<<dim3(H / 128, (int)(BS / 128)), 256, 0, stream>>>(
      ao, woT, out, (int)BS, H, H);
}
